// MSDeformableAttention_14293651161297
// MI455X (gfx1250) — matrix operand order repaired and checked by reading, not compiled, not run on hardware
//
#include <hip/hip_runtime.h>
#include <hip/hip_bf16.h>

// ---------------------------------------------------------------------------
// MS-Deformable-Attention for gfx1250 (MI455X).
//
//   1) f32 -> f16 convert of the 4 weight matrices (tiny)
//   2) WMMA GEMM (A=f32 value, fused cvt): vH  = value @ vp_w + vp_b  -> f16
//   3) WMMA GEMM (A=f32 query):            so  = query @ so_w + so_b  -> f32
//   4) WMMA GEMM (A=f32 query):            awl = query @ aw_w + aw_b  -> f32
//   5) sampler: inline softmax(16) + bilinear gather from f16 v (54MB,
//      L2-resident), lane==channel (hd==32==wave32), writes t as f16
//   6) WMMA GEMM (A=f16 t):                out = t @ op_w + op_b      -> f32
//
// GEMM: wave tile 32x64 (two A tiles share every B load), operands loaded
// before the 8-WMMA group so loadcnt waits can be partial.
// ---------------------------------------------------------------------------

typedef __attribute__((ext_vector_type(16))) _Float16 v16h;
typedef __attribute__((ext_vector_type(8)))  _Float16 v8h;
typedef __attribute__((ext_vector_type(8)))  float    v8f;

#define BS      8
#define LEN_Q   1024
#define EMBED   256
#define NHEAD   8
#define NLVL    4
#define NPTS    4
#define HDIM    32
#define LEN_V   13294
#define KDIM    256   // K for every GEMM in this model

// ---------------------------------------------------------------------------
// f32 -> f16 elementwise convert (weights only)
// ---------------------------------------------------------------------------
__global__ void f32_to_f16_kernel(const float* __restrict__ src,
                                  _Float16* __restrict__ dst, int n) {
    int i = blockIdx.x * blockDim.x + threadIdx.x;
    if (i < n) dst[i] = (_Float16)src[i];
}

// ---------------------------------------------------------------------------
// A-tile loaders: produce the 16x32 f16 A-operand per-lane layout
// (CDNA5 ISA 7.12.2): lane L holds row M=L%16;
//   halves 0..7  -> K = k +  8*(L/16)      (k = 0..7)
//   halves 8..15 -> K = k + 16 + 8*(L/16)  (k = 0..7)
// ---------------------------------------------------------------------------
__device__ __forceinline__ v16h load_a_tile(const _Float16* __restrict__ p, int g) {
    v8h lo = *(const v8h*)(p + 8 * g);
    v8h hi = *(const v8h*)(p + 16 + 8 * g);
    v16h a;
#pragma unroll
    for (int i = 0; i < 8; ++i) { a[i] = lo[i]; a[i + 8] = hi[i]; }
    return a;
}
__device__ __forceinline__ v16h load_a_tile(const float* __restrict__ p, int g) {
    v8f lo = *(const v8f*)(p + 8 * g);          // fused f32 load + cvt
    v8f hi = *(const v8f*)(p + 16 + 8 * g);
    v16h a;
#pragma unroll
    for (int i = 0; i < 8; ++i) { a[i] = (_Float16)lo[i]; a[i + 8] = (_Float16)hi[i]; }
    return a;
}

__device__ __forceinline__ void store_c(float* C, size_t i, float v)     { C[i] = v; }
__device__ __forceinline__ void store_c(_Float16* C, size_t i, float v)  { C[i] = (_Float16)v; }

// ---------------------------------------------------------------------------
// WMMA GEMM:  C[M,N] = A[M,K] @ B[K,N] (f16, row-major) + bias[N]
// AT in {float (cvt on load), _Float16}; CT in {float, _Float16}.
// K == KDIM (256), N % 64 == 0, M arbitrary (clamped loads, guarded stores).
// Block = 256 threads = 8 waves; wave owns 32 rows x 64 cols
// (block: 256 rows x 64 cols). 8 WMMAs per K-block per wave.
// B per-lane layout: lane L holds column N=L%16 of the 32x16 tile, 16 strided
// K halves in the same k order as the A tile (K = 8g+i, 16+8g+i-8).
// ---------------------------------------------------------------------------
template <typename AT, typename CT>
__global__ void wmma_gemm_bias_kernel(const AT* __restrict__ A,
                                      const _Float16* __restrict__ B,
                                      const float* __restrict__ bias,
                                      CT* __restrict__ C,
                                      int M, int N) {
    const int lane  = threadIdx.x & 31;
    const int wave  = threadIdx.x >> 5;
    const int g     = lane >> 4;                 // half-wave group
    const int mBase = blockIdx.y * 256 + wave * 32;
    const int nBase = blockIdx.x * 64;

    int r0 = mBase + (lane & 15);      r0 = r0 < M ? r0 : (M - 1);
    int r1 = mBase + 16 + (lane & 15); r1 = r1 < M ? r1 : (M - 1);
    const AT* __restrict__ aRow0 = A + (size_t)r0 * KDIM;
    const AT* __restrict__ aRow1 = A + (size_t)r1 * KDIM;

    v8f acc[2][4];
#pragma unroll
    for (int s = 0; s < 2; ++s)
#pragma unroll
        for (int t = 0; t < 4; ++t)
#pragma unroll
            for (int r = 0; r < 8; ++r) acc[s][t][r] = 0.0f;

#pragma unroll 2
    for (int kb = 0; kb < KDIM; kb += 32) {
        // ---- issue all operand loads for this K-block up front ----
        v16h a0 = load_a_tile(aRow0 + kb, g);
        v16h a1 = load_a_tile(aRow1 + kb, g);
        __builtin_prefetch(aRow0 + kb + 64, 0, 0);   // next K-slab of streamed A
        __builtin_prefetch(aRow1 + kb + 64, 0, 0);

        const _Float16* __restrict__ bCol = B + (size_t)kb * N + nBase + (lane & 15);
        v16h b0, b1, b2, b3;
#pragma unroll
        for (int i = 0; i < 16; ++i) {
            const size_t ko = (size_t)(i < 8 ? 8 * g + i : 16 + 8 * g + (i - 8)) * N;
            b0[i] = bCol[ko];
            b1[i] = bCol[ko + 16];
            b2[i] = bCol[ko + 32];
            b3[i] = bCol[ko + 48];
        }

        // ---- 8 WMMAs: two A tiles reuse each B register set ----
        acc[0][0] = __builtin_amdgcn_wmma_f32_16x16x32_f16(false, a0, false, b0, (short)0, acc[0][0], false, false);
        acc[1][0] = __builtin_amdgcn_wmma_f32_16x16x32_f16(false, a1, false, b0, (short)0, acc[1][0], false, false);
        acc[0][1] = __builtin_amdgcn_wmma_f32_16x16x32_f16(false, a0, false, b1, (short)0, acc[0][1], false, false);
        acc[1][1] = __builtin_amdgcn_wmma_f32_16x16x32_f16(false, a1, false, b1, (short)0, acc[1][1], false, false);
        acc[0][2] = __builtin_amdgcn_wmma_f32_16x16x32_f16(false, a0, false, b2, (short)0, acc[0][2], false, false);
        acc[1][2] = __builtin_amdgcn_wmma_f32_16x16x32_f16(false, a1, false, b2, (short)0, acc[1][2], false, false);
        acc[0][3] = __builtin_amdgcn_wmma_f32_16x16x32_f16(false, a0, false, b3, (short)0, acc[0][3], false, false);
        acc[1][3] = __builtin_amdgcn_wmma_f32_16x16x32_f16(false, a1, false, b3, (short)0, acc[1][3], false, false);
    }

    // ---- bias + store (C layout: M = r + 8g (+16s), N = lane%16) ----
#pragma unroll
    for (int t = 0; t < 4; ++t) {
        const int col = nBase + 16 * t + (lane & 15);
        const float bv = bias[col];
#pragma unroll
        for (int s = 0; s < 2; ++s)
#pragma unroll
            for (int r = 0; r < 8; ++r) {
                const int row = mBase + 16 * s + r + 8 * g;
                if (row < M) store_c(C, (size_t)row * N + col, acc[s][t][r] + bv);
            }
    }
}

// ---------------------------------------------------------------------------
// Sampling kernel: one wave per (b,q,h); lane == channel (HDIM == 32).
// Softmax over the 16 (l,p) logits computed redundantly per lane (uniform,
// broadcast loads). v is f16 (54MB -> resident in the 192MB L2); each corner
// gather is a coalesced 64B wave read. Writes t as f16 for the final GEMM.
// ---------------------------------------------------------------------------
__global__ void msda_sample_kernel(const float* __restrict__ so,
                                   const float* __restrict__ awl,
                                   const float* __restrict__ ref,
                                   const _Float16* __restrict__ v,
                                   _Float16* __restrict__ t) {
    const int c = threadIdx.x;       // channel 0..31
    const int h = threadIdx.y;       // head    0..7
    const int q = blockIdx.x;
    const int b = blockIdx.y;
    const size_t bq = (size_t)b * LEN_Q + q;

    constexpr int lvlH[4] = {100, 50, 25, 13};
    constexpr int lvlW[4] = {100, 50, 25, 13};
    constexpr int lvlS[4] = {0, 10000, 12500, 13125};

    const float* __restrict__ soP = so  + bq * (NHEAD * NLVL * NPTS * 2) + h * (NLVL * NPTS * 2);
    const float* __restrict__ awP = awl + bq * (NHEAD * NLVL * NPTS)     + h * (NLVL * NPTS);

    // softmax over 16 logits (wave-uniform)
    float lg[16];
    float mx = -1e30f;
#pragma unroll
    for (int i = 0; i < 16; ++i) { lg[i] = awP[i]; mx = fmaxf(mx, lg[i]); }
    float s = 0.0f;
#pragma unroll
    for (int i = 0; i < 16; ++i) { lg[i] = __expf(lg[i] - mx); s += lg[i]; }
    const float inv = 1.0f / s;

    const _Float16* __restrict__ vb = v + (size_t)b * LEN_V * EMBED + h * HDIM + c;

    float acc = 0.0f;
#pragma unroll
    for (int l = 0; l < NLVL; ++l) {
        const int lh = lvlH[l], lw = lvlW[l];
        const float rx = ref[(bq * NLVL + l) * 2 + 0];
        const float ry = ref[(bq * NLVL + l) * 2 + 1];
        const _Float16* __restrict__ vl = vb + (size_t)lvlS[l] * EMBED;
        const float fw = (float)lw, fh = (float)lh;
#pragma unroll
        for (int p = 0; p < NPTS; ++p) {
            const float ox = soP[(l * NPTS + p) * 2 + 0];
            const float oy = soP[(l * NPTS + p) * 2 + 1];
            // x = ((2*loc-1)+1)*W/2 - 0.5 = loc*W - 0.5
            const float x = (rx + ox / fw) * fw - 0.5f;
            const float y = (ry + oy / fh) * fh - 0.5f;
            const float xf = floorf(x), yf = floorf(y);
            const int   x0 = (int)xf,   y0 = (int)yf;
            const float fx = x - xf,    fy = y - yf;
            const float w  = lg[l * NPTS + p] * inv;

            const bool xv0 = (x0 >= 0) & (x0 < lw);
            const bool xv1 = (x0 + 1 >= 0) & (x0 + 1 < lw);
            const bool yv0 = (y0 >= 0) & (y0 < lh);
            const bool yv1 = (y0 + 1 >= 0) & (y0 + 1 < lh);

            float v00 = 0.f, v01 = 0.f, v10 = 0.f, v11 = 0.f;
            if (yv0 & xv0) v00 = (float)vl[(size_t)(y0 * lw + x0) * EMBED];
            if (yv0 & xv1) v01 = (float)vl[(size_t)(y0 * lw + x0 + 1) * EMBED];
            if (yv1 & xv0) v10 = (float)vl[(size_t)((y0 + 1) * lw + x0) * EMBED];
            if (yv1 & xv1) v11 = (float)vl[(size_t)((y0 + 1) * lw + x0 + 1) * EMBED];

            const float bil = (1.0f - fy) * ((1.0f - fx) * v00 + fx * v01)
                            +          fy * ((1.0f - fx) * v10 + fx * v11);
            acc = fmaf(w, bil, acc);
        }
    }
    t[bq * EMBED + h * HDIM + c] = (_Float16)acc;
}

// ---------------------------------------------------------------------------
// Host-side orchestration
// ---------------------------------------------------------------------------
extern "C" void kernel_launch(void* const* d_in, const int* in_sizes, int n_in,
                              void* d_out, int out_size, void* d_ws, size_t ws_size,
                              hipStream_t stream) {
    (void)in_sizes; (void)n_in; (void)out_size; (void)ws_size;

    // inputs in setup_inputs() order
    const float* query = (const float*)d_in[0];   // [8,1024,256]
    const float* refp  = (const float*)d_in[1];   // [8,1024,4,2]
    const float* value = (const float*)d_in[2];   // [8,13294,256]
    // d_in[3] = spatial shapes (int64) -- constants hardcoded
    const float* so_w  = (const float*)d_in[4];   // [256,256]
    const float* so_b  = (const float*)d_in[5];   // [256]
    const float* aw_w  = (const float*)d_in[6];   // [256,128]
    const float* aw_b  = (const float*)d_in[7];   // [128]
    const float* vp_w  = (const float*)d_in[8];   // [256,256]
    const float* vp_b  = (const float*)d_in[9];   // [256]
    const float* op_w  = (const float*)d_in[10];  // [256,256]
    const float* op_b  = (const float*)d_in[11];  // [256]
    float* out = (float*)d_out;                   // [8,1024,256]

    const int MV = BS * LEN_V;   // 106352 rows of value
    const int MQ = BS * LEN_Q;   // 8192 rows of query

    // workspace carve-up (256B aligned)
    char*  ws  = (char*)d_ws;
    size_t off = 0;
    auto carve = [&](size_t bytes) {
        size_t o = off;
        off = (off + bytes + 255) & ~(size_t)255;
        return o;
    };
    _Float16* vpwH = (_Float16*)(ws + carve((size_t)EMBED * EMBED * 2));
    _Float16* sowH = (_Float16*)(ws + carve((size_t)EMBED * EMBED * 2));
    _Float16* awwH = (_Float16*)(ws + carve((size_t)EMBED * 128 * 2));
    _Float16* opwH = (_Float16*)(ws + carve((size_t)EMBED * EMBED * 2));
    _Float16* vH   = (_Float16*)(ws + carve((size_t)MV * EMBED * 2));
    float*    soF  = (float*)   (ws + carve((size_t)MQ * EMBED * 4));
    float*    awF  = (float*)   (ws + carve((size_t)MQ * 128 * 4));
    _Float16* tH   = (_Float16*)(ws + carve((size_t)MQ * EMBED * 2));

    // 1) convert the four weight matrices to f16
    auto cvt = [&](const float* s, _Float16* d, int n) {
        f32_to_f16_kernel<<<dim3((n + 255) / 256), dim3(256), 0, stream>>>(s, d, n);
    };
    cvt(vp_w, vpwH, EMBED * EMBED);
    cvt(so_w, sowH, EMBED * EMBED);
    cvt(aw_w, awwH, EMBED * 128);
    cvt(op_w, opwH, EMBED * EMBED);

    // 2) v = value @ vp_w + vp_b   (106352 x 256, f32 A fused-cvt, f16 out)
    wmma_gemm_bias_kernel<float, _Float16>
        <<<dim3(EMBED / 64, (MV + 255) / 256), dim3(256), 0, stream>>>(
            value, vpwH, vp_b, vH, MV, EMBED);

    // 3) so = query @ so_w + so_b  (8192 x 256, f32 out)
    wmma_gemm_bias_kernel<float, float>
        <<<dim3(EMBED / 64, MQ / 256), dim3(256), 0, stream>>>(
            query, sowH, so_b, soF, MQ, EMBED);

    // 4) awl = query @ aw_w + aw_b (8192 x 128, f32 out)
    wmma_gemm_bias_kernel<float, float>
        <<<dim3(128 / 64, MQ / 256), dim3(256), 0, stream>>>(
            query, awwH, aw_b, awF, MQ, 128);

    // 5) softmax + bilinear sampling -> t (f16)
    msda_sample_kernel<<<dim3(LEN_Q, BS), dim3(HDIM, NHEAD), 0, stream>>>(
        soF, awF, refp, vH, tH);

    // 6) out = t @ op_w + op_b     (8192 x 256, f16 A, f32 out)
    wmma_gemm_bias_kernel<_Float16, float>
        <<<dim3(EMBED / 64, MQ / 256), dim3(256), 0, stream>>>(
            tH, opwH, op_b, out, MQ, EMBED);
}
